// LocalAttentionBlock_57019985822495
// MI455X (gfx1250) — matrix operand order repaired and checked by reading, not compiled, not run on hardware
//
#include <hip/hip_runtime.h>
#include <cstdint>

typedef __bf16 bf16;
typedef __attribute__((ext_vector_type(16))) __bf16 v16bf;
typedef __attribute__((ext_vector_type(8)))  __bf16 v8bf;
typedef __attribute__((ext_vector_type(8)))  float  v8f;
typedef __attribute__((ext_vector_type(4)))  float  f32x4;
typedef unsigned int u32x4 __attribute__((ext_vector_type(4)));
typedef int          i32x8 __attribute__((ext_vector_type(8)));
typedef int          i32x4 __attribute__((ext_vector_type(4)));

#define B_   2
#define S_   2048
#define DM_  1024
#define H_   16
#define HD_  64
#define FF_  4096
#define NT_  (B_*S_)          // 4096 token rows
#define SCALE_ 0.125f         // 1/sqrt(64)
#define EPS_  1e-5f

// ---------------------------------------------------------------- WMMA helper
__device__ __forceinline__ v8f wmma_bf16f32(v16bf a, v16bf b, v8f c) {
  return __builtin_amdgcn_wmma_f32_16x16x32_bf16(
      /*neg_a=*/false, a, /*neg_b=*/false, b,
      /*c_mod=*/(short)0, c, /*reuse_a=*/false, /*reuse_b=*/false);
}

// A-matrix 16x32 bf16 fragment element index (per ISA 7.12.2), i = vgpr pair
__device__ __forceinline__ int a_kidx(int i, int hi) {
  return (i < 4) ? (2*i + 8*hi) : (16 + 2*(i-4) + 8*hi);
}

// ----------------------------------------------------- TDM descriptor + issue
// 2D tile DMA: global (row-major, stride_elems bf16 per row) -> LDS, with
// optional LDS row padding (pad codes are in DWORD units per D# spec).
__device__ __forceinline__ uint32_t lds_off(const void* p) {
  return (uint32_t)(uintptr_t)p;     // LDS aperture: addr[31:0] = LDS offset
}

__device__ __forceinline__ void tdm_load_2d(
    uint32_t lds_addr, const bf16* gptr,
    uint32_t tile_d0, uint32_t tile_d1, uint64_t stride_elems,
    uint32_t pad_interval, uint32_t pad_amount) {
  uint64_t ga = (uint64_t)(uintptr_t)gptr;
  u32x4 g0;
  g0[0] = 1u;                                          // count=1 (valid D#)
  g0[1] = lds_addr;                                    // lds_addr (bytes)
  g0[2] = (uint32_t)ga;                                // global_addr[31:0]
  g0[3] = (uint32_t)((ga >> 32) & 0x1ffffffu) | (2u << 30);  // addr[56:32]|type=2
  const uint32_t td0 = 1u << 30, td1 = 1u << 30;       // huge dims: no OOB clip
  i32x8 g1;
  g1[0] = (int)((1u << 16)                 // data_size = 2 bytes
              | (1u << 20)                 // pad_enable
              | (pad_interval << 22)
              | (pad_amount  << 25));
  g1[1] = (int)((td0 & 0xffffu) << 16);                // tensor_dim0[15:0]
  g1[2] = (int)((td0 >> 16) | ((td1 & 0xffffu) << 16));// dim0 hi | dim1 lo
  g1[3] = (int)((td1 >> 16) | (tile_d0 << 16));        // dim1 hi | tile_dim0
  g1[4] = (int)(tile_d1 & 0xffffu);                    // tile_dim1 (tile_dim2=0)
  g1[5] = (int)(uint32_t)(stride_elems & 0xffffffffu); // dim0_stride[31:0]
  g1[6] = (int)(uint32_t)((stride_elems >> 32) & 0xffffu);
  g1[7] = 0;
  i32x4 z4 = {0, 0, 0, 0};                             // 2D: groups 2/3 unused
  i32x8 z8 = {0, 0, 0, 0, 0, 0, 0, 0};
  __builtin_amdgcn_tensor_load_to_lds(g0, g1, z4, z4, z8, 0);
}

// ---------------------------------------------------------------- utils
__global__ void cast_f32_bf16_kernel(const float* __restrict__ in,
                                     bf16* __restrict__ out, size_t n) {
  size_t i = (size_t)blockIdx.x * blockDim.x + threadIdx.x;
  size_t stride = (size_t)gridDim.x * blockDim.x;
  for (; i < n; i += stride) out[i] = (bf16)in[i];
}

__global__ void zero_f32_kernel(f32x4* __restrict__ p, size_t n4) {
  size_t i = (size_t)blockIdx.x * blockDim.x + threadIdx.x;
  size_t stride = (size_t)gridDim.x * blockDim.x;
  f32x4 z = {0.f, 0.f, 0.f, 0.f};
  for (; i < n4; i += stride) __builtin_nontemporal_store(z, &p[i]);
}

// ------------------------------------------------- fused LayerNorm -> bf16
__global__ __launch_bounds__(256) void ln_bf16_kernel(
    const float* __restrict__ x, const float* __restrict__ gamma,
    const float* __restrict__ beta, bf16* __restrict__ out) {
  __shared__ float red[8];
  int row = blockIdx.x;
  const float* xr = x + (size_t)row * DM_;
  float v0[4];
  float s = 0.f;
#pragma unroll
  for (int j = 0; j < 4; ++j) { v0[j] = xr[threadIdx.x + 256*j]; s += v0[j]; }
#pragma unroll
  for (int off = 16; off > 0; off >>= 1) s += __shfl_xor(s, off, 32);
  int wid = threadIdx.x >> 5, lane = threadIdx.x & 31;
  if (lane == 0) red[wid] = s;
  __syncthreads();
  float tot = 0.f;
#pragma unroll
  for (int w = 0; w < 8; ++w) tot += red[w];
  float mean = tot * (1.0f / DM_);
  __syncthreads();
  float sq = 0.f;
#pragma unroll
  for (int j = 0; j < 4; ++j) { float d = v0[j] - mean; sq += d * d; }
#pragma unroll
  for (int off = 16; off > 0; off >>= 1) sq += __shfl_xor(sq, off, 32);
  if (lane == 0) red[wid] = sq;
  __syncthreads();
  float vt = 0.f;
#pragma unroll
  for (int w = 0; w < 8; ++w) vt += red[w];
  float rstd = rsqrtf(vt * (1.0f / DM_) + EPS_);
  bf16* orow = out + (size_t)row * DM_;
#pragma unroll
  for (int j = 0; j < 4; ++j) {
    int c = threadIdx.x + 256*j;
    orow[c] = (bf16)((v0[j] - mean) * rstd * gamma[c] + beta[c]);
  }
}

// --------------------------------------------------------- tiled WMMA GEMM
// C[M,N] = A[M,K](bf16) @ W[K,N](bf16) + bias, epilogue:
//  MODE 0: -> bf16   MODE 1: +residual -> f32   MODE 2: GELU -> bf16
// LDS tiles are double-buffered and filled by the Tensor Data Mover
// (wave 0 issues tensor_load_to_lds, waits on TENSORcnt, barrier releases).
#define BM 128
#define BN 128
#define BK 32
#define LDA_P 40    // bf16 elems; 80B row stride  (32 data + 8 pad = 16+4 DW)
#define LDB_P 136   // bf16 elems; 272B row stride (128 data + 8 pad = 64+4 DW)
#define PADI_A 3    // pad after 16 DWORDs
#define PADI_B 5    // pad after 64 DWORDs
#define PADA_4 3    // pad amount 4 DWORDs

template <int MODE>
__global__ __launch_bounds__(256) void gemm_bf16_kernel(
    const bf16* __restrict__ A, const bf16* __restrict__ W,
    const float* __restrict__ bias, const float* __restrict__ res,
    bf16* __restrict__ outb, float* __restrict__ outf,
    int M, int N, int K) {
  __shared__ bf16 sA[2][BM * LDA_P];
  __shared__ bf16 sB[2][BK * LDB_P];

  int tid = threadIdx.x;
  int lane = tid & 31, wid = tid >> 5;
  int wm = wid & 3, wn = wid >> 2;          // wave grid 4x2 over 128x128
  int lm = lane & 15, hi = lane >> 4;
  int bMr = blockIdx.y * BM, bNc = blockIdx.x * BN;

  v8f acc[2][4] = {};

  const bf16* Abase = A + (size_t)bMr * K;
  const bf16* Wbase = W + bNc;

  if (wid == 0) {   // prologue: DMA tile 0 into buffer 0
    tdm_load_2d(lds_off(&sA[0][0]), Abase, BK, BM, (uint64_t)K, PADI_A, PADA_4);
    tdm_load_2d(lds_off(&sB[0][0]), Wbase, BN, BK, (uint64_t)N, PADI_B, PADA_4);
  }

  int cur = 0;
  for (int k0 = 0; k0 < K; k0 += BK) {
    int nxt = cur ^ 1;
    if (wid == 0) {
      if (k0 + BK < K) {   // DMA next tile while current computes
        tdm_load_2d(lds_off(&sA[nxt][0]), Abase + (k0 + BK),
                    BK, BM, (uint64_t)K, PADI_A, PADA_4);
        tdm_load_2d(lds_off(&sB[nxt][0]), Wbase + (size_t)(k0 + BK) * N,
                    BN, BK, (uint64_t)N, PADI_B, PADA_4);
        __builtin_amdgcn_s_wait_tensorcnt(2);   // current tile's pair landed
      } else {
        __builtin_amdgcn_s_wait_tensorcnt(0);
      }
    }
    __syncthreads();   // wave0 arrives only after tensor data is in LDS

    v16bf af[2], bfr[4];
#pragma unroll
    for (int mt = 0; mt < 2; ++mt) {
      const bf16* base = &sA[cur][(wm * 32 + mt * 16 + lm) * LDA_P];
#pragma unroll
      for (int i = 0; i < 8; ++i) {
        int kk = a_kidx(i, hi);
        af[mt][2*i]   = base[kk];
        af[mt][2*i+1] = base[kk + 1];
      }
    }
#pragma unroll
    for (int nt = 0; nt < 4; ++nt) {
      const bf16* bb = &sB[cur][wn * 64 + nt * 16 + lm];
#pragma unroll
      for (int i = 0; i < 8; ++i) {
        int kk = a_kidx(i, hi);
        bfr[nt][2*i]   = bb[(size_t)kk * LDB_P];
        bfr[nt][2*i+1] = bb[(size_t)(kk + 1) * LDB_P];
      }
    }
#pragma unroll
    for (int mt = 0; mt < 2; ++mt)
#pragma unroll
      for (int nt = 0; nt < 4; ++nt)
        acc[mt][nt] = wmma_bf16f32(af[mt], bfr[nt], acc[mt][nt]);
    __syncthreads();   // done reading buffer before TDM overwrites it
    cur = nxt;
  }

#pragma unroll
  for (int mt = 0; mt < 2; ++mt) {
#pragma unroll
    for (int nt = 0; nt < 4; ++nt) {
      int col = bNc + wn * 64 + nt * 16 + lm;
      float bv = bias[col];
#pragma unroll
      for (int r = 0; r < 8; ++r) {
        int row = bMr + wm * 32 + mt * 16 + r + 8 * hi;
        size_t o = (size_t)row * N + col;
        float val = acc[mt][nt][r] + bv;
        if (MODE == 1) {
          outf[o] = val + res[o];
        } else if (MODE == 2) {
          outb[o] = (bf16)(0.5f * val * (1.0f + erff(val * 0.70710678118f)));
        } else {
          outb[o] = (bf16)val;
        }
      }
    }
  }
}

// ------------------------------------------------- banded local attention
// Block = 64 threads (2 waves); each wave owns one 16-query tile.
// Keys covered per tile: [qt-128, qt+159] = 288 = 18 tiles (band is ±128).
#define KTILES 18
#define KEYS   (KTILES * 16)   // 288

__global__ __launch_bounds__(64) void attn_kernel(
    const bf16* __restrict__ q, const bf16* __restrict__ k,
    const bf16* __restrict__ v, float* __restrict__ attn,
    bf16* __restrict__ ctx) {
  __shared__ float sS[2][16][KEYS];   // 36 KB scores
  __shared__ bf16  sP[2][16][KEYS];   // 18 KB probabilities (bf16)

  int lane = threadIdx.x & 31, wid = threadIdx.x >> 5;
  int lm = lane & 15, hi = lane >> 4;
  int idx = blockIdx.x;               // b*H*(S/32) + h*(S/32) + q32
  int q32 = idx & 63;
  int h   = (idx >> 6) & (H_ - 1);
  int b   = idx >> 10;
  int qt  = q32 * 32 + wid * 16;      // query tile start
  int key0 = qt - 128;                // first (possibly negative) key

  const bf16* qbase = q + ((size_t)b * S_ + qt) * DM_ + h * HD_;

  // Q A-fragments for the two 32-deep K-steps of HD=64
  v16bf aq[2];
#pragma unroll
  for (int s = 0; s < 2; ++s) {
    const bf16* qr = qbase + (size_t)lm * DM_ + s * 32;
#pragma unroll
    for (int i = 0; i < 8; ++i) {
      int kk = a_kidx(i, hi);
      aq[s][2*i]   = qr[kk];
      aq[s][2*i+1] = qr[kk + 1];
    }
  }

  // Pass 1: scores (Q·K^T), masked+scaled into LDS
  for (int t = 0; t < KTILES; ++t) {
    int kb = key0 + t * 16;
    int key = min(max(kb + lm, 0), S_ - 1);
    const bf16* kr = k + ((size_t)b * S_ + key) * DM_ + h * HD_;
    v8f c = {};
#pragma unroll
    for (int s = 0; s < 2; ++s) {
      v16bf bk;
#pragma unroll
      for (int i = 0; i < 8; ++i) {
        int kk = 32 * s + a_kidx(i, hi);
        bk[2*i]   = kr[kk];
        bk[2*i+1] = kr[kk + 1];
      }
      c = wmma_bf16f32(aq[s], bk, c);
    }
#pragma unroll
    for (int r = 0; r < 8; ++r) {
      int m = r + 8 * hi;
      int i = qt + m;
      int j = kb + lm;
      bool ok = (j >= 0) && (j < S_) && (abs(i - j) <= 128);
      sS[wid][m][t * 16 + lm] = ok ? c[r] * SCALE_ : -INFINITY;
    }
  }
  __syncthreads();

  // Pass 2: row softmax; band probabilities streamed (non-temporal) to attn
  if (lane < 16) {
    int m = lane;
    float mx = -INFINITY;
    for (int n = 0; n < KEYS; ++n) mx = fmaxf(mx, sS[wid][m][n]);
    float sum = 0.f;
    for (int n = 0; n < KEYS; ++n) sum += __expf(sS[wid][m][n] - mx);
    float inv = 1.0f / sum;
    int i = qt + m;
    float* arow = attn + (((size_t)(b * H_ + h) * S_ + i) * S_);
    for (int n = 0; n < KEYS; ++n) {
      float p = __expf(sS[wid][m][n] - mx) * inv;
      sP[wid][m][n] = (bf16)p;
      int j = key0 + n;
      if (j >= 0 && j < S_) __builtin_nontemporal_store(p, &arow[j]);
    }
  }
  __syncthreads();

  // Pass 3: context = P @ V  (K-dim = 288 keys, N = HD=64)
  v8f acc[4] = {};
  for (int ks = 0; ks < KEYS / 32; ++ks) {
    v16bf ap;
    const bf16* pr = &sP[wid][lm][ks * 32];
#pragma unroll
    for (int i = 0; i < 8; ++i) {
      int kk = a_kidx(i, hi);
      ap[2*i]   = pr[kk];
      ap[2*i+1] = pr[kk + 1];
    }
#pragma unroll
    for (int nt = 0; nt < 4; ++nt) {
      v16bf bvf;
      const bf16* vcol = v + (size_t)b * S_ * DM_ + h * HD_ + nt * 16 + lm;
#pragma unroll
      for (int i = 0; i < 8; ++i) {
        int kk = a_kidx(i, hi);
        int k1 = min(max(key0 + ks * 32 + kk, 0), S_ - 1);
        int k2 = min(max(key0 + ks * 32 + kk + 1, 0), S_ - 1);
        bvf[2*i]   = vcol[(size_t)k1 * DM_];
        bvf[2*i+1] = vcol[(size_t)k2 * DM_];
      }
      acc[nt] = wmma_bf16f32(ap, bvf, acc[nt]);
    }
  }
#pragma unroll
  for (int nt = 0; nt < 4; ++nt)
#pragma unroll
    for (int r = 0; r < 8; ++r) {
      int i = qt + r + 8 * hi;
      ctx[((size_t)b * S_ + i) * DM_ + h * HD_ + nt * 16 + lm] = (bf16)acc[nt][r];
    }
}

// ---------------------------------------------------------------- launcher
extern "C" void kernel_launch(void* const* d_in, const int* in_sizes, int n_in,
                              void* d_out, int out_size, void* d_ws, size_t ws_size,
                              hipStream_t stream) {
  (void)in_sizes; (void)n_in; (void)out_size; (void)ws_size;
  const float* x     = (const float*)d_in[0];
  const float* ln1s  = (const float*)d_in[1];
  const float* ln1b  = (const float*)d_in[2];
  const float* wq    = (const float*)d_in[3];
  const float* bq    = (const float*)d_in[4];
  const float* wk    = (const float*)d_in[5];
  const float* bk    = (const float*)d_in[6];
  const float* wv    = (const float*)d_in[7];
  const float* bv    = (const float*)d_in[8];
  const float* wo    = (const float*)d_in[9];
  const float* bo    = (const float*)d_in[10];
  const float* ln2s  = (const float*)d_in[11];
  const float* ln2b  = (const float*)d_in[12];
  const float* w1    = (const float*)d_in[13];
  const float* b1    = (const float*)d_in[14];
  const float* w2    = (const float*)d_in[15];
  const float* b2    = (const float*)d_in[16];

  char* ws = (char*)d_ws;
  const size_t MB = 1ull << 20;
  bf16* xn   = (bf16*)(ws + 0);        // 8MB (reused for xn2)
  bf16* qb   = (bf16*)(ws + 8  * MB);  // 8MB
  bf16* kb   = (bf16*)(ws + 16 * MB);  // 8MB
  bf16* vb   = (bf16*)(ws + 24 * MB);  // 8MB
  bf16* ctx  = (bf16*)(ws + 32 * MB);  // 8MB
  bf16* hbuf = (bf16*)(ws + 8  * MB);  // 32MB, reuses q/k/v/ctx after attention
  float* x1  = (float*)(ws + 40 * MB); // 16MB
  bf16* wqb  = (bf16*)(ws + 56 * MB);
  bf16* wkb  = (bf16*)(ws + 58 * MB);
  bf16* wvb  = (bf16*)(ws + 60 * MB);
  bf16* wob  = (bf16*)(ws + 62 * MB);
  bf16* w1b  = (bf16*)(ws + 64 * MB);  // 8MB
  bf16* w2b  = (bf16*)(ws + 72 * MB);  // 8MB -> total 80MB

  float* out_x    = (float*)d_out;                       // (B,S,DM)
  float* out_attn = out_x + (size_t)B_ * S_ * DM_;       // (B,H,S,S)

  // 1) cast weights to bf16
  cast_f32_bf16_kernel<<<1024, 256, 0, stream>>>(wq, wqb, (size_t)DM_ * DM_);
  cast_f32_bf16_kernel<<<1024, 256, 0, stream>>>(wk, wkb, (size_t)DM_ * DM_);
  cast_f32_bf16_kernel<<<1024, 256, 0, stream>>>(wv, wvb, (size_t)DM_ * DM_);
  cast_f32_bf16_kernel<<<1024, 256, 0, stream>>>(wo, wob, (size_t)DM_ * DM_);
  cast_f32_bf16_kernel<<<2048, 256, 0, stream>>>(w1, w1b, (size_t)DM_ * FF_);
  cast_f32_bf16_kernel<<<2048, 256, 0, stream>>>(w2, w2b, (size_t)FF_ * DM_);

  // 2) zero the full attn output, non-temporal (band overwritten later)
  size_t attn_n4 = (size_t)B_ * H_ * S_ * S_ / 4;
  zero_f32_kernel<<<8192, 256, 0, stream>>>((f32x4*)out_attn, attn_n4);

  // 3) LN1 -> xn (bf16)
  ln_bf16_kernel<<<NT_, 256, 0, stream>>>(x, ln1s, ln1b, xn);

  // 4) Q/K/V projections
  dim3 gP(DM_ / BN, NT_ / BM);
  gemm_bf16_kernel<0><<<gP, 256, 0, stream>>>(xn, wqb, bq, nullptr, qb, nullptr, NT_, DM_, DM_);
  gemm_bf16_kernel<0><<<gP, 256, 0, stream>>>(xn, wkb, bk, nullptr, kb, nullptr, NT_, DM_, DM_);
  gemm_bf16_kernel<0><<<gP, 256, 0, stream>>>(xn, wvb, bv, nullptr, vb, nullptr, NT_, DM_, DM_);

  // 5) banded attention -> attn (f32) + ctx (bf16)
  attn_kernel<<<B_ * H_ * (S_ / 32), 64, 0, stream>>>(qb, kb, vb, out_attn, ctx);

  // 6) output projection + residual -> x1 (f32)
  gemm_bf16_kernel<1><<<gP, 256, 0, stream>>>(ctx, wob, bo, x, nullptr, x1, NT_, DM_, DM_);

  // 7) LN2 -> xn2 (reuse xn)
  ln_bf16_kernel<<<NT_, 256, 0, stream>>>(x1, ln2s, ln2b, xn);

  // 8) FFN1 + GELU -> hbuf (bf16)
  dim3 gF1(FF_ / BN, NT_ / BM);
  gemm_bf16_kernel<2><<<gF1, 256, 0, stream>>>(xn, w1b, b1, nullptr, hbuf, nullptr, NT_, FF_, DM_);

  // 9) FFN2 + residual -> out x (f32)
  gemm_bf16_kernel<1><<<gP, 256, 0, stream>>>(hbuf, w2b, b2, x1, nullptr, out_x, NT_, DM_, FF_);
}
